// InteractionLayer_9560597201472
// MI455X (gfx1250) — matrix operand order repaired and checked by reading, not compiled, not run on hardware
//
#include <hip/hip_runtime.h>
#include <cstdint>

typedef __attribute__((ext_vector_type(16))) _Float16 v16h;
typedef __attribute__((ext_vector_type(8)))  _Float16 v8h;
typedef __attribute__((ext_vector_type(2)))  _Float16 v2h;
typedef __attribute__((ext_vector_type(8)))  float    v8f;

#define HIDDEN 128
#define NBASES 64
#define D_MAXV 25.0f
#define EWAVES 8
#define NWAVES 4

__device__ __forceinline__ v16h cat8(v8h lo, v8h hi) {
  return __builtin_shufflevector(lo, hi, 0,1,2,3,4,5,6,7,8,9,10,11,12,13,14,15);
}

// Native fp32 global atomic add, no return (STOREcnt), device scope.
__device__ __forceinline__ void atomic_add_f32_dev(float* p, float v) {
  asm volatile("global_atomic_add_f32 %0, %1, off scope:SCOPE_DEV"
               :: "v"(p), "v"(v) : "memory");
}

// B fragment for v_wmma_f32_16x16x32_f16: B is 32x16 (K x N), K-tile kt.
// Weights stored transposed: [N][Ktot] halves, K contiguous.
// Lane l holds column N = l%16; halves: K = kt*32 + sel + {0..7} and + {16..23}, sel = (l<16?0:8).
__device__ __forceinline__ v16h load_bfrag(const _Float16* Wt, int Ktot, int n, int kt, int lane) {
  int off = n * Ktot + kt * 32 + ((lane & 16) ? 8 : 0);
  v8h lo = *(const v8h*)(Wt + off);
  v8h hi = *(const v8h*)(Wt + off + 16);
  return cat8(lo, hi);
}

// A fragment: A is 16x32 (M x K). Activations stored row-major [16][Ktot] halves.
// Lane l holds row M = l%16; halves: K = kt*32 + sel + {0..7} and + {16..23}, sel = (l<16?0:8).
__device__ __forceinline__ v16h load_afrag(const _Float16* Act, int Ktot, int kt, int lane) {
  int m   = lane & 15;
  int sel = (lane & 16) ? 8 : 0;
  int off = m * Ktot + kt * 32 + sel;
  v8h lo = *(const v8h*)(Act + off);
  v8h hi = *(const v8h*)(Act + off + 16);
  return cat8(lo, hi);
}

__device__ __forceinline__ v8f wmma16(v16h a, v16h b, v8f c) {
  return __builtin_amdgcn_wmma_f32_16x16x32_f16(false, a, false, b, (short)0, c, false, false);
}

// ---------------------------------------------------------------- prep
__global__ void prep_kernel(const float* __restrict__ W1, const float* __restrict__ W2,
                            const float* __restrict__ U1, const float* __restrict__ U2,
                            _Float16* __restrict__ W1T, _Float16* __restrict__ W2T,
                            _Float16* __restrict__ U1T, _Float16* __restrict__ U2T) {
  int t = blockIdx.x * blockDim.x + threadIdx.x;
  int stride = gridDim.x * blockDim.x;
  for (int i = t; i < NBASES * HIDDEN; i += stride) {
    int k = i / HIDDEN, n = i % HIDDEN;
    W1T[n * NBASES + k] = (_Float16)W1[i];
  }
  for (int i = t; i < HIDDEN * HIDDEN; i += stride) {
    int k = i / HIDDEN, n = i % HIDDEN;
    W2T[n * HIDDEN + k] = (_Float16)W2[i];
    U1T[n * HIDDEN + k] = (_Float16)U1[i];
    U2T[n * HIDDEN + k] = (_Float16)U2[i];
  }
}

__global__ void zero_kernel(float* __restrict__ H, int n) {
  int t = blockIdx.x * blockDim.x + threadIdx.x;
  int stride = gridDim.x * blockDim.x;
  for (int i = t; i < n; i += stride) H[i] = 0.0f;
}

// ---------------------------------------------------------------- edge kernel
// One wave per 16-edge tile: RBF -> WMMA layer1 -> ReLU -> WMMA layer2 -> ReLU
// -> X[src]*M scatter-add into H[dest].
__global__ void __launch_bounds__(EWAVES * 32) edge_kernel(
    const float* __restrict__ X, const float* __restrict__ R,
    const int* __restrict__ src, const int* __restrict__ dest,
    const _Float16* __restrict__ gW1T, const _Float16* __restrict__ gW2T,
    float* __restrict__ H, int numTiles) {
  __shared__ alignas(16) _Float16 sW1T[HIDDEN * NBASES];
  __shared__ alignas(16) _Float16 sW2T[HIDDEN * HIDDEN];
  __shared__ alignas(16) _Float16 sAct[EWAVES][16 * HIDDEN];

  { // cooperative weight staging into LDS (128-bit copies)
    const uint4* s1 = (const uint4*)gW1T; uint4* d1 = (uint4*)sW1T;
    for (int i = threadIdx.x; i < HIDDEN * NBASES / 8; i += blockDim.x) d1[i] = s1[i];
    const uint4* s2 = (const uint4*)gW2T; uint4* d2 = (uint4*)sW2T;
    for (int i = threadIdx.x; i < HIDDEN * HIDDEN / 8; i += blockDim.x) d2[i] = s2[i];
  }
  __syncthreads();

  const int wave = threadIdx.x >> 5;
  const int lane = threadIdx.x & 31;
  _Float16* act = sAct[wave];
  const float width  = D_MAXV / (NBASES - 1);
  const float inv2w2 = 1.0f / (2.0f * width * width);

  for (int tile = blockIdx.x * EWAVES + wave; tile < numTiles; tile += gridDim.x * EWAVES) {
    int m = lane & 15;
    int e = tile * 16 + m;
    int s = src[e];
    int d = dest[e];
    float dx = R[s * 3 + 0] - R[d * 3 + 0];
    float dy = R[s * 3 + 1] - R[d * 3 + 1];
    float dz = R[s * 3 + 2] - R[d * 3 + 2];
    float dist = dx * dx + dy * dy + dz * dz;

    // Build RBF A-fragments (K tiles 0 and 1) directly in registers.
    int sel = (lane & 16) ? 8 : 0;
    v16h a0, a1;
#pragma unroll
    for (int h = 0; h < 16; ++h) {
      int kk = sel + (h < 8 ? h : h + 8);
      float t0 = dist - (float)kk * width;
      float t1 = dist - (float)(kk + 32) * width;
      a0[h] = (_Float16)__expf(-t0 * t0 * inv2w2);
      a1[h] = (_Float16)__expf(-t1 * t1 * inv2w2);
    }

    // Layer 1: rbf[16x64] @ W1[64x128], ReLU, to LDS (layout C->A transpose).
#pragma unroll
    for (int nt = 0; nt < 8; ++nt) {
      int n = nt * 16 + (lane & 15);
      v8f c = {};
      c = wmma16(a0, load_bfrag(sW1T, NBASES, n, 0, lane), c);
      c = wmma16(a1, load_bfrag(sW1T, NBASES, n, 1, lane), c);
#pragma unroll
      for (int i = 0; i < 8; ++i) {
        int mr = i + ((lane & 16) ? 8 : 0);
        act[mr * HIDDEN + n] = (_Float16)fmaxf(c[i], 0.0f);
      }
    }
    __builtin_amdgcn_wave_barrier(); // LDS in-order per wave; stop reordering

    // Layer 2: act[16x128] @ W2[128x128], ReLU, then gather-multiply-scatter.
#pragma unroll
    for (int nt = 0; nt < 8; ++nt) {
      int n = nt * 16 + (lane & 15);
      v8f c = {};
#pragma unroll
      for (int kt = 0; kt < 4; ++kt)
        c = wmma16(load_afrag(act, HIDDEN, kt, lane),
                   load_bfrag(sW2T, HIDDEN, n, kt, lane), c);
#pragma unroll
      for (int i = 0; i < 8; ++i) {
        int mr = i + ((lane & 16) ? 8 : 0);
        float f  = fmaxf(c[i], 0.0f);
        int   sm = __shfl(s, mr, 32);
        int   dm = __shfl(d, mr, 32);
        float val = f * X[(size_t)sm * HIDDEN + n];
        atomic_add_f32_dev(&H[(size_t)dm * HIDDEN + n], val);
      }
    }
    __builtin_amdgcn_wave_barrier();
  }
}

// ---------------------------------------------------------------- node kernel
// out = relu(H @ U1 + b1) @ U2 + b2, 16 rows per wave.
__global__ void __launch_bounds__(NWAVES * 32) node_kernel(
    const float* __restrict__ H,
    const _Float16* __restrict__ gU1T, const _Float16* __restrict__ gU2T,
    const float* __restrict__ b1, const float* __restrict__ b2,
    float* __restrict__ out, int numTiles) {
  __shared__ alignas(16) _Float16 sU1T[HIDDEN * HIDDEN];
  __shared__ alignas(16) _Float16 sU2T[HIDDEN * HIDDEN];
  __shared__ alignas(16) _Float16 sIn[NWAVES][16 * HIDDEN];
  __shared__ alignas(16) _Float16 sAct[NWAVES][16 * HIDDEN];

  {
    const uint4* s1 = (const uint4*)gU1T; uint4* d1 = (uint4*)sU1T;
    for (int i = threadIdx.x; i < HIDDEN * HIDDEN / 8; i += blockDim.x) d1[i] = s1[i];
    const uint4* s2 = (const uint4*)gU2T; uint4* d2 = (uint4*)sU2T;
    for (int i = threadIdx.x; i < HIDDEN * HIDDEN / 8; i += blockDim.x) d2[i] = s2[i];
  }
  __syncthreads();

  const int wave = threadIdx.x >> 5;
  const int lane = threadIdx.x & 31;
  _Float16* in  = sIn[wave];
  _Float16* act = sAct[wave];

  for (int tile = blockIdx.x * NWAVES + wave; tile < numTiles; tile += gridDim.x * NWAVES) {
    // Stage 16x128 f32 tile of H into LDS as f16 (coalesced float2 loads).
    const float2* Hp = (const float2*)(H + (size_t)tile * 16 * HIDDEN);
#pragma unroll
    for (int it = 0; it < 32; ++it) {
      int idx = it * 32 + lane;        // 1024 float2 per tile
      float2 v = Hp[idx];
      v2h h = { (_Float16)v.x, (_Float16)v.y };
      *(v2h*)(in + idx * 2) = h;
    }
    __builtin_amdgcn_wave_barrier();

    // Layer 1: relu(H @ U1 + b1)
#pragma unroll
    for (int nt = 0; nt < 8; ++nt) {
      int n = nt * 16 + (lane & 15);
      v8f c = {};
#pragma unroll
      for (int kt = 0; kt < 4; ++kt)
        c = wmma16(load_afrag(in, HIDDEN, kt, lane),
                   load_bfrag(sU1T, HIDDEN, n, kt, lane), c);
      float bias = b1[n];
#pragma unroll
      for (int i = 0; i < 8; ++i) {
        int mr = i + ((lane & 16) ? 8 : 0);
        act[mr * HIDDEN + n] = (_Float16)fmaxf(c[i] + bias, 0.0f);
      }
    }
    __builtin_amdgcn_wave_barrier();

    // Layer 2: (.) @ U2 + b2 -> fp32 out
#pragma unroll
    for (int nt = 0; nt < 8; ++nt) {
      int n = nt * 16 + (lane & 15);
      v8f c = {};
#pragma unroll
      for (int kt = 0; kt < 4; ++kt)
        c = wmma16(load_afrag(act, HIDDEN, kt, lane),
                   load_bfrag(sU2T, HIDDEN, n, kt, lane), c);
      float bias = b2[n];
#pragma unroll
      for (int i = 0; i < 8; ++i) {
        int mr = i + ((lane & 16) ? 8 : 0);
        out[((size_t)tile * 16 + mr) * HIDDEN + n] = c[i] + bias;
      }
    }
    __builtin_amdgcn_wave_barrier();
  }
}

// ---------------------------------------------------------------- launch
extern "C" void kernel_launch(void* const* d_in, const int* in_sizes, int n_in,
                              void* d_out, int out_size, void* d_ws, size_t ws_size,
                              hipStream_t stream) {
  const float* X    = (const float*)d_in[0];
  const float* R    = (const float*)d_in[1];
  const int*   src  = (const int*)d_in[2];
  const int*   dest = (const int*)d_in[3];
  const float* W1   = (const float*)d_in[4];
  const float* W2   = (const float*)d_in[5];
  const float* U1   = (const float*)d_in[6];
  const float* b1   = (const float*)d_in[7];
  const float* U2   = (const float*)d_in[8];
  const float* b2   = (const float*)d_in[9];
  float* out = (float*)d_out;

  const int V = in_sizes[0] / HIDDEN;   // 50000
  const int E = in_sizes[2];            // 800000

  char* ws = (char*)d_ws;
  float* H = (float*)ws;
  size_t off = (size_t)V * HIDDEN * sizeof(float);
  off = (off + 255) & ~(size_t)255;
  _Float16* W1T = (_Float16*)(ws + off); off += (size_t)NBASES * HIDDEN * 2;
  _Float16* W2T = (_Float16*)(ws + off); off += (size_t)HIDDEN * HIDDEN * 2;
  _Float16* U1T = (_Float16*)(ws + off); off += (size_t)HIDDEN * HIDDEN * 2;
  _Float16* U2T = (_Float16*)(ws + off); off += (size_t)HIDDEN * HIDDEN * 2;

  hipLaunchKernelGGL(prep_kernel, dim3(64), dim3(256), 0, stream,
                     W1, W2, U1, U2, W1T, W2T, U1T, U2T);
  hipLaunchKernelGGL(zero_kernel, dim3(2048), dim3(256), 0, stream, H, V * HIDDEN);

  int etiles = E / 16;
  hipLaunchKernelGGL(edge_kernel, dim3(512), dim3(EWAVES * 32), 0, stream,
                     X, R, src, dest, W1T, W2T, H, etiles);

  int ntiles = V / 16;
  hipLaunchKernelGGL(node_kernel, dim3(512), dim3(NWAVES * 32), 0, stream,
                     H, U1T, U2T, b1, b2, out, ntiles);
}
